// WaveNet_89507118449260
// MI455X (gfx1250) — compile-verified, matrix operand order checked by reading, not run on hardware
//
#include <hip/hip_runtime.h>
#include <hip/hip_bf16.h>

// ---------------------------------------------------------------------------
// WaveNet forward for MI455X (gfx1250), wave32, bf16 WMMA everywhere.
//   prep:   f32 weights -> bf16 WMMA-A layouts; sum skip biases
//   embed:  one-hot conv == embedding gather -> h0 [B,32,T] f32
//   20x layer: fg = Wc(64x64) @ [h(t-d);h(t)]  (WMMA)
//              z  = tanh(f)*sigmoid(g)         (v_tanh_f32 / v_exp_f32)
//              h' = h + Wr(32x32) @ z          (WMMA)   -> ping-pong h
//   head:   skip = relu(Ws_all(256x640) @ z_all + bs_sum)  (WMMA, K chunked,
//                  z chunks staged via async global->LDS b64 when available)
//           mid  = relu(We1(256x256) @ skip + be1)          (WMMA)
//           out  = We2(256x256) @ mid + be2                 (WMMA) -> d_out
// ---------------------------------------------------------------------------

typedef __attribute__((ext_vector_type(16))) __bf16 v16bf;
typedef __attribute__((ext_vector_type(8)))  float  v8f;
typedef int wn_v2i __attribute__((__vector_size__(2 * sizeof(int))));

#define T_LEN 16384
#define N_B   4
#define N_LAY 20

#if defined(__gfx1250__) && \
    __has_builtin(__builtin_amdgcn_global_load_async_to_lds_b64) && \
    __has_builtin(__builtin_amdgcn_s_wait_asynccnt)
#define WN_ASYNC 1
#define AS1 __attribute__((address_space(1)))
#define AS3 __attribute__((address_space(3)))
#else
#define WN_ASYNC 0
#endif

__device__ __forceinline__ unsigned short f2bf(float f) {
    union { float f; unsigned u; } v; v.f = f;
    unsigned r = v.u + 0x7FFFu + ((v.u >> 16) & 1u);   // round-to-nearest-even
    return (unsigned short)(r >> 16);
}

__device__ __forceinline__ float fast_rcp(float x) {
#if __has_builtin(__builtin_amdgcn_rcpf)
    return __builtin_amdgcn_rcpf(x);
#else
    return 1.f / x;
#endif
}

__device__ __forceinline__ float fast_tanh(float x) {
#if __has_builtin(__builtin_amdgcn_tanhf)
    return __builtin_amdgcn_tanhf(x);          // gfx1250 v_tanh_f32 (TRANS)
#else
    float xc = fminf(fmaxf(x, -9.f), 9.f);     // clamp so exp can't overflow
    float e  = __expf(2.f * xc);
    return (e - 1.f) * fast_rcp(e + 1.f);
#endif
}

__device__ __forceinline__ float fast_sigmoid(float x) {
    return fast_rcp(1.f + __expf(-x));
}

// Per-lane fragment gather for row-major [m][k] (A) or [n][k] (B in LDS).
// ISA 7.12.2 16-bit layout: lanes 0-15 hold k {0..7,16..23} (+8 for lanes
// 16-31), two bf16 packed per VGPR -> 8 dword loads (2 contiguous 16B runs).
__device__ __forceinline__ v16bf load_frag(const unsigned short* base, int row0,
                                           int ldk, int kBase, int lane) {
    int r  = row0 + (lane & 15);
    int hh = lane >> 4;
    union { v16bf v; unsigned u[8]; } f;
#pragma unroll
    for (int j = 0; j < 8; ++j) {
        int k = kBase + ((j < 4) ? (hh * 8 + j * 2) : (16 + hh * 8 + (j - 4) * 2));
        f.u[j] = *reinterpret_cast<const unsigned*>(base + r * ldk + k);
    }
    return f.v;
}

// Same fragment from a k-major [k][n] LDS tile (used for async-staged chunks):
// 16 u16 reads; compiler pairs them into d16/d16_hi loads.
__device__ __forceinline__ v16bf load_frag_kmajor(const unsigned short* base, int n0,
                                                  int ldn, int kBase, int lane) {
    int n  = n0 + (lane & 15);
    int hh = lane >> 4;
    union { v16bf v; unsigned u[8]; } f;
#pragma unroll
    for (int j = 0; j < 8; ++j) {
        int k = kBase + ((j < 4) ? (hh * 8 + j * 2) : (16 + hh * 8 + (j - 4) * 2));
        unsigned lo = base[k * ldn + n];
        unsigned hi = base[(k + 1) * ldn + n];
        f.u[j] = lo | (hi << 16);
    }
    return f.v;
}

__device__ __forceinline__ v8f wmma_bf16(v16bf a, v16bf b, v8f c) {
    return __builtin_amdgcn_wmma_f32_16x16x32_bf16(false, a, false, b,
                                                   (short)0, c, false, false);
}

// ---------------------------------------------------------------------------
// prep: weight conversion to bf16 WMMA-A layouts + bias pre-sum
// ---------------------------------------------------------------------------
__global__ __launch_bounds__(256) void wn_prep(
    const float* __restrict__ Wc, const float* __restrict__ Wr,
    const float* __restrict__ Ws, const float* __restrict__ We1,
    const float* __restrict__ We2, const float* __restrict__ bs,
    unsigned short* __restrict__ WcT, unsigned short* __restrict__ WrB,
    unsigned short* __restrict__ WsB, unsigned short* __restrict__ We1B,
    unsigned short* __restrict__ We2B, float* __restrict__ bs_sum) {
    int gid = blockIdx.x * blockDim.x + threadIdx.x;
    int stride = gridDim.x * blockDim.x;
    // Wc[20][64][32][2] -> WcT[i][o][k], k = tap*32 + c  (tap0 = t-d, tap1 = t)
    for (int idx = gid; idx < N_LAY * 64 * 64; idx += stride) {
        int k = idx & 63, o = (idx >> 6) & 63, i = idx >> 12;
        int tap = k >> 5, c = k & 31;
        WcT[idx] = f2bf(Wc[((i * 64 + o) * 32 + c) * 2 + tap]);
    }
    for (int idx = gid; idx < N_LAY * 32 * 32; idx += stride)
        WrB[idx] = f2bf(Wr[idx]);                       // [i][o][c]
    // Ws[20][256][32] -> WsB[s][i*32+c]
    for (int idx = gid; idx < 256 * 640; idx += stride) {
        int kk = idx % 640, s = idx / 640;
        int i = kk >> 5, c = kk & 31;
        WsB[idx] = f2bf(Ws[(i * 256 + s) * 32 + c]);
    }
    for (int idx = gid; idx < 256 * 256; idx += stride) We1B[idx] = f2bf(We1[idx]);
    for (int idx = gid; idx < 256 * 256; idx += stride) We2B[idx] = f2bf(We2[idx]);
    for (int idx = gid; idx < 256; idx += stride) {
        float s = 0.f;
        for (int i = 0; i < N_LAY; ++i) s += bs[i * 256 + idx];
        bs_sum[idx] = s;
    }
}

// ---------------------------------------------------------------------------
// embed: one-hot causal conv (K=2,d=1) == 2-tap embedding lookup -> h f32
// ---------------------------------------------------------------------------
__global__ __launch_bounds__(256) void wn_embed(
    const int* __restrict__ x, const float* __restrict__ W0,
    const float* __restrict__ b0, float* __restrict__ h) {
    int t = blockIdx.x * blockDim.x + threadIdx.x;
    int b = blockIdx.y;
    int xc = x[b * T_LEN + t];
    int xp = (t > 0) ? x[b * T_LEN + t - 1] : -1;
#pragma unroll 4
    for (int c = 0; c < 32; ++c) {
        float v = W0[(c * 256 + xc) * 2 + 1] + b0[c];
        if (xp >= 0) v += W0[(c * 256 + xp) * 2 + 0];
        h[(b * 32 + c) * T_LEN + t] = v;
    }
}

// ---------------------------------------------------------------------------
// layer: dilated gated conv + residual, one 64-timestep tile per workgroup
// ---------------------------------------------------------------------------
__global__ __launch_bounds__(256) void wn_layer(
    const float* __restrict__ h_in, float* __restrict__ h_out,
    unsigned short* __restrict__ z_all,
    const unsigned short* __restrict__ WcT, const unsigned short* __restrict__ WrB,
    const float* __restrict__ bc, const float* __restrict__ br,
    int layer, int dil) {
    int b = blockIdx.y;
    int t0 = blockIdx.x * 64;
    int tid = threadIdx.x, lane = tid & 31, wv = tid >> 5;

    __shared__ unsigned short Bconv[64][64];   // [tt][k] k<32: h(t-d), k>=32: h(t)
    __shared__ float          FG[64][65];      // [o][tt] conv accum, padded
    __shared__ unsigned short Bz[64][32];      // [tt][c] gated output, bf16

    // Phase A: stage conv inputs (coalesced along t)
#pragma unroll
    for (int it = 0; it < 16; ++it) {
        int idx = it * 256 + tid;
        int tt = idx & 63, k = idx >> 6, c = k & 31;
        int t = t0 + tt - ((k < 32) ? dil : 0);
        float v = (t >= 0) ? h_in[(b * 32 + c) * T_LEN + t] : 0.f;
        Bconv[tt][k] = f2bf(v);
    }
    __syncthreads();

    // Phase B: fg(64 x 64) = WcT(64x64) @ Bconv  -- 16 tiles, 2 per wave
    const unsigned short* Wl = WcT + layer * 4096;
#pragma unroll
    for (int q = 0; q < 2; ++q) {
        int tile = wv * 2 + q;
        int mt = tile >> 2, nt = tile & 3;
        v8f acc = {0.f, 0.f, 0.f, 0.f, 0.f, 0.f, 0.f, 0.f};
#pragma unroll
        for (int ks = 0; ks < 2; ++ks) {
            v16bf a  = load_frag(Wl, mt * 16, 64, ks * 32, lane);
            v16bf bb = load_frag(&Bconv[0][0], nt * 16, 64, ks * 32, lane);
            acc = wmma_bf16(a, bb, acc);
        }
        int n = lane & 15, hh = lane >> 4;
#pragma unroll
        for (int r = 0; r < 8; ++r)
            FG[mt * 16 + hh * 8 + r][nt * 16 + n] = acc[r];
    }
    __syncthreads();

    // Phase C: z = tanh(f)*sigmoid(g); write bf16 to LDS + z_all
#pragma unroll
    for (int it = 0; it < 8; ++it) {
        int idx = it * 256 + tid;
        int tt = idx & 63, c = idx >> 6;
        float f = FG[c][tt]      + bc[layer * 64 + c];
        float g = FG[c + 32][tt] + bc[layer * 64 + 32 + c];
        float z = fast_tanh(f) * fast_sigmoid(g);
        unsigned short zb = f2bf(z);
        Bz[tt][c] = zb;
        z_all[((layer * N_B + b) * 32 + c) * T_LEN + t0 + tt] = zb;
    }
    __syncthreads();

    // Phase D: h' = h + Wr(32x32) @ z + br  -- 8 tiles, 1 per wave
    {
        int mt = wv >> 2, nt = wv & 3;
        v8f acc = {0.f, 0.f, 0.f, 0.f, 0.f, 0.f, 0.f, 0.f};
        v16bf a  = load_frag(WrB + layer * 1024, mt * 16, 32, 0, lane);
        v16bf bb = load_frag(&Bz[0][0], nt * 16, 32, 0, lane);
        acc = wmma_bf16(a, bb, acc);
        int n = lane & 15, hh = lane >> 4;
#pragma unroll
        for (int r = 0; r < 8; ++r) {
            int m = mt * 16 + hh * 8 + r;
            int t = t0 + nt * 16 + n;
            h_out[(b * 32 + m) * T_LEN + t] =
                acc[r] + h_in[(b * 32 + m) * T_LEN + t] + br[layer * 32 + m];
        }
    }
}

// ---------------------------------------------------------------------------
// head: skip = relu(Ws_all @ z_all + bs_sum); relu(We1 @ . + be1); We2 @ . + be2
// 64-timestep tile per WG; skip K=640 chunked 4x160 so LDS stays at 64 KB.
// z chunk staged k-major with 8-byte copies (async global->LDS when available).
// ---------------------------------------------------------------------------
__global__ __launch_bounds__(256) void wn_head(
    const unsigned short* __restrict__ z_all, const unsigned short* __restrict__ WsB,
    const unsigned short* __restrict__ We1B, const unsigned short* __restrict__ We2B,
    const float* __restrict__ bs_sum, const float* __restrict__ be1,
    const float* __restrict__ be2, float* __restrict__ out) {
    int b = blockIdx.y;
    int t0 = blockIdx.x * 64;
    int tid = threadIdx.x, lane = tid & 31, wv = tid >> 5;
    int n = lane & 15, hh = lane >> 4;

    __shared__ char smem[65536];
    unsigned short* Zk = (unsigned short*)smem;             // [160][64] k-major
    unsigned short* S1 = (unsigned short*)(smem + 32768);   // [64][256] skip bf16
    unsigned short* S2 = (unsigned short*)smem;             // [64][256] mid  bf16

    v8f acc[8];
#pragma unroll
    for (int i = 0; i < 8; ++i)
        acc[i] = (v8f){0.f, 0.f, 0.f, 0.f, 0.f, 0.f, 0.f, 0.f};

    // ---- skip = Ws_all(256x640) @ z_all, K chunked (5 layers per chunk) ----
    for (int cc = 0; cc < 4; ++cc) {
        // Stage chunk: 2560 lane-copies of 8 bytes (4 bf16 along t), k-major.
#pragma unroll
        for (int it = 0; it < 10; ++it) {
            int idx  = it * 256 + tid;
            int sub  = idx & 15;             // 4-timestep group
            int kloc = idx >> 4;             // 0..159
            int kk = cc * 160 + kloc;
            int lay = kk >> 5, c = kk & 31;
            const unsigned short* g =
                z_all + (size_t)((lay * N_B + b) * 32 + c) * T_LEN + t0 + sub * 4;
            unsigned short* l = Zk + kloc * 64 + sub * 4;
#if WN_ASYNC
            __builtin_amdgcn_global_load_async_to_lds_b64(
                (AS1 wn_v2i*)(__UINTPTR_TYPE__)g,
                (AS3 wn_v2i*)(unsigned)(__UINTPTR_TYPE__)l, 0, 0);
#else
            *reinterpret_cast<unsigned long long*>(l) =
                *reinterpret_cast<const unsigned long long*>(g);
#endif
        }
#if WN_ASYNC
        __builtin_amdgcn_s_wait_asynccnt(0);
#endif
        __syncthreads();
#pragma unroll
        for (int ti = 0; ti < 8; ++ti) {
            int tile = wv * 8 + ti;
            int mt = tile >> 2, nt = tile & 3;
#pragma unroll
            for (int ks = 0; ks < 5; ++ks) {
                v16bf a  = load_frag(WsB, mt * 16, 640, cc * 160 + ks * 32, lane);
                v16bf bb = load_frag_kmajor(Zk, nt * 16, 64, ks * 32, lane);
                acc[ti] = wmma_bf16(a, bb, acc[ti]);
            }
        }
        __syncthreads();
    }
#pragma unroll
    for (int ti = 0; ti < 8; ++ti) {
        int tile = wv * 8 + ti;
        int mt = tile >> 2, nt = tile & 3;
#pragma unroll
        for (int r = 0; r < 8; ++r) {
            int s = mt * 16 + hh * 8 + r;
            float v = fmaxf(acc[ti][r] + bs_sum[s], 0.f);
            S1[(nt * 16 + n) * 256 + s] = f2bf(v);
        }
    }
    __syncthreads();

    // ---- mid = relu(We1 @ skip + be1) ----
#pragma unroll
    for (int ti = 0; ti < 8; ++ti)
        acc[ti] = (v8f){0.f, 0.f, 0.f, 0.f, 0.f, 0.f, 0.f, 0.f};
#pragma unroll
    for (int ti = 0; ti < 8; ++ti) {
        int tile = wv * 8 + ti;
        int mt = tile >> 2, nt = tile & 3;
#pragma unroll
        for (int ks = 0; ks < 8; ++ks) {
            v16bf a  = load_frag(We1B, mt * 16, 256, ks * 32, lane);
            v16bf bb = load_frag(S1, nt * 16, 256, ks * 32, lane);
            acc[ti] = wmma_bf16(a, bb, acc[ti]);
        }
    }
#pragma unroll
    for (int ti = 0; ti < 8; ++ti) {   // S2 region disjoint from S1 -> safe
        int tile = wv * 8 + ti;
        int mt = tile >> 2, nt = tile & 3;
#pragma unroll
        for (int r = 0; r < 8; ++r) {
            int m = mt * 16 + hh * 8 + r;
            float v = fmaxf(acc[ti][r] + be1[m], 0.f);
            S2[(nt * 16 + n) * 256 + m] = f2bf(v);
        }
    }
    __syncthreads();

    // ---- out = We2 @ mid + be2 ----
#pragma unroll
    for (int ti = 0; ti < 8; ++ti)
        acc[ti] = (v8f){0.f, 0.f, 0.f, 0.f, 0.f, 0.f, 0.f, 0.f};
#pragma unroll
    for (int ti = 0; ti < 8; ++ti) {
        int tile = wv * 8 + ti;
        int mt = tile >> 2, nt = tile & 3;
#pragma unroll
        for (int ks = 0; ks < 8; ++ks) {
            v16bf a  = load_frag(We2B, mt * 16, 256, ks * 32, lane);
            v16bf bb = load_frag(S2, nt * 16, 256, ks * 32, lane);
            acc[ti] = wmma_bf16(a, bb, acc[ti]);
        }
    }
#pragma unroll
    for (int ti = 0; ti < 8; ++ti) {
        int tile = wv * 8 + ti;
        int mt = tile >> 2, nt = tile & 3;
#pragma unroll
        for (int r = 0; r < 8; ++r) {
            int q = mt * 16 + hh * 8 + r;
            int t = t0 + nt * 16 + n;
            out[(b * 256 + q) * T_LEN + t] = acc[ti][r] + be2[q];
        }
    }
}

// ---------------------------------------------------------------------------
extern "C" void kernel_launch(void* const* d_in, const int* in_sizes, int n_in,
                              void* d_out, int out_size, void* d_ws, size_t ws_size,
                              hipStream_t stream) {
    (void)in_sizes; (void)n_in; (void)out_size; (void)ws_size;
    const int*   x   = (const int*)d_in[0];
    const float* W0  = (const float*)d_in[1];
    const float* b0  = (const float*)d_in[2];
    const float* Wc  = (const float*)d_in[3];
    const float* bc  = (const float*)d_in[4];
    const float* Wr  = (const float*)d_in[5];
    const float* br  = (const float*)d_in[6];
    const float* Ws  = (const float*)d_in[7];
    const float* bs  = (const float*)d_in[8];
    const float* We1 = (const float*)d_in[9];
    const float* be1 = (const float*)d_in[10];
    const float* We2 = (const float*)d_in[11];
    const float* be2 = (const float*)d_in[12];
    float* out = (float*)d_out;

    char* ws = (char*)d_ws;
    size_t off = 0;
    auto alloc = [&](size_t bytes) {
        char* p = ws + off;
        off = (off + bytes + 255) & ~(size_t)255;
        return p;
    };
    float*          hA    = (float*)alloc((size_t)N_B * 32 * T_LEN * 4);
    float*          hB    = (float*)alloc((size_t)N_B * 32 * T_LEN * 4);
    unsigned short* z_all = (unsigned short*)alloc((size_t)N_LAY * N_B * 32 * T_LEN * 2);
    unsigned short* WcT   = (unsigned short*)alloc(N_LAY * 64 * 64 * 2);
    unsigned short* WrB   = (unsigned short*)alloc(N_LAY * 32 * 32 * 2);
    unsigned short* WsB   = (unsigned short*)alloc(256 * 640 * 2);
    unsigned short* We1B  = (unsigned short*)alloc(256 * 256 * 2);
    unsigned short* We2B  = (unsigned short*)alloc(256 * 256 * 2);
    float*          bsSum = (float*)alloc(256 * 4);

    wn_prep<<<dim3(128), dim3(256), 0, stream>>>(Wc, Wr, Ws, We1, We2, bs,
                                                 WcT, WrB, WsB, We1B, We2B, bsSum);
    wn_embed<<<dim3(T_LEN / 256, N_B), dim3(256), 0, stream>>>(x, W0, b0, hA);

    static const int DIL[N_LAY] = {1, 2, 4, 8, 16, 32, 64, 128, 256, 512,
                                   1, 2, 4, 8, 16, 32, 64, 128, 256, 512};
    float* hin = hA;
    float* hout = hB;
    for (int i = 0; i < N_LAY; ++i) {
        wn_layer<<<dim3(T_LEN / 64, N_B), dim3(256), 0, stream>>>(
            hin, hout, z_all, WcT, WrB, bc, br, i, DIL[i]);
        float* tmp = hin; hin = hout; hout = tmp;
    }
    wn_head<<<dim3(T_LEN / 64, N_B), dim3(256), 0, stream>>>(
        z_all, WsB, We1B, We2B, bsSum, be1, be2, out);
}
